// DGAConv_46926812676544
// MI455X (gfx1250) — matrix operand order repaired and checked by reading, not compiled, not run on hardware
//
#include <hip/hip_runtime.h>
#include <hip/hip_bf16.h>
#include <cstdint>

// ---------------------------------------------------------------------------
// Problem constants (match reference)
// ---------------------------------------------------------------------------
#define NN   4096      // nodes
#define CC   256       // channels
#define EE   131072    // edges
#define NC   (NN*CC)   // 1,048,576 elements per [N,C] buffer

// GEMM tiling
#define GTM 128
#define GTN 64
#define GTK 32
#define ASTR 40        // LDS stride (elements) for A tile rows  -> 80B, 16B-aligned
#define BSTR 40        // LDS stride (elements) for Bt tile rows

typedef __attribute__((ext_vector_type(16))) __bf16 v16bf;
typedef __attribute__((ext_vector_type(8)))  float  v8f;

union FragU { uint4 u[2]; v16bf v; };

// round-to-nearest-even f32 -> bf16 (bit pattern)
__device__ __forceinline__ unsigned short f2bf(float f) {
    unsigned int u = __float_as_uint(f);
    unsigned int r = u + 0x7fffu + ((u >> 16) & 1u);
    return (unsigned short)(r >> 16);
}

// two f32 -> packed 2x bf16 in one dword (RNE), no sub-word register traffic
__device__ __forceinline__ unsigned int pack_bf16(float f0, float f1) {
    unsigned int u0 = __float_as_uint(f0);
    unsigned int u1 = __float_as_uint(f1);
    u0 = u0 + 0x7fffu + ((u0 >> 16) & 1u);
    u1 = u1 + 0x7fffu + ((u1 >> 16) & 1u);
    return (u0 >> 16) | (u1 & 0xffff0000u);
}

__device__ __forceinline__ float act_apply(float x, int act) {
    switch (act) {
        case 1: return fmaxf(x, 0.0f);                          // relu
        case 2: return 1.0f / (1.0f + __expf(-x));              // sigmoid
        case 3: return tanhf(x);                                // tanh
        case 4: {                                               // gelu (tanh approx, jax default)
            float x3 = x * x * x;
            return 0.5f * x * (1.0f + tanhf(0.7978845608028654f * (x + 0.044715f * x3)));
        }
        default: return x;
    }
}

// ---------------------------------------------------------------------------
// Generic WMMA bf16 GEMM:  Out[M,Nc] (f32) = epilogue( A[M,K] @ B[K,Nc] )
// A,B are f32 in memory; converted to bf16 while staging into LDS.
// ta=1: A is stored [K,M] and used transposed (A^T @ ... access A[k*M+m]).
// Epilogue: v = act(acc + bias[col]); if(mul) v*=mul[idx]; if(res) v+=res[idx];
//           accum ? Out[idx]+=v : Out[idx]=v
// Requires: M % 128 == 0, Nc % 64 == 0, K % 32 == 0 (true for all calls here).
// ---------------------------------------------------------------------------
__global__ __launch_bounds__(256) void gemm_wmma_bf16(
    const float* __restrict__ A, const float* __restrict__ B,
    float* __restrict__ Out, int M, int K, int Nc, int ta,
    const float* __restrict__ bias, int act,
    const float* __restrict__ mulBuf, const float* __restrict__ resBuf, int accum)
{
    __shared__ __align__(16) unsigned short As[GTM * ASTR]; // [m][k]
    __shared__ __align__(16) unsigned short Bs[GTN * BSTR]; // [n][k] (transposed)

    const int tid  = threadIdx.x;
    const int m0   = blockIdx.x * GTM;
    const int n0   = blockIdx.y * GTN;
    const int lane = tid & 31;
    const int wave = tid >> 5;      // 8 waves
    const int wm   = wave >> 1;     // 0..3 -> 32-row strip
    const int wn   = wave & 1;      // 0..1 -> 32-col strip

    v8f acc[2][2];
    #pragma unroll
    for (int i = 0; i < 2; ++i)
        #pragma unroll
        for (int j = 0; j < 2; ++j)
            #pragma unroll
            for (int r = 0; r < 8; ++r) acc[i][j][r] = 0.0f;

    for (int k0 = 0; k0 < K; k0 += GTK) {
        // ---------------- stage A tile ----------------
        if (!ta) {
            // row-major A: thread -> (row, 16-col half); 16 consecutive f32 per thread
            int r  = tid >> 1;       // 0..127
            int hh = tid & 1;        // 0..1
            const float* ap = A + (size_t)(m0 + r) * K + k0 + hh * 16;
            if (k0 + GTK < K)
                __builtin_prefetch((const void*)(ap + GTK), 0, 0);  // global_prefetch_b8
            unsigned int* dst = (unsigned int*)&As[r * ASTR + hh * 16];
            #pragma unroll
            for (int j = 0; j < 4; ++j) {
                float4 f = ((const float4*)ap)[j];
                dst[2*j+0] = pack_bf16(f.x, f.y);
                dst[2*j+1] = pack_bf16(f.z, f.w);
            }
        } else {
            // A stored [K,M]; tile element (r,c) = A[(k0+c)*M + (m0+r)]
            int c   = tid >> 3;      // 0..31  (k within tile)
            int seg = tid & 7;       // 0..7   (16-row chunk)
            const float* ap = A + (size_t)(k0 + c) * M + m0 + seg * 16;
            #pragma unroll
            for (int j = 0; j < 16; ++j)
                As[(seg * 16 + j) * ASTR + c] = f2bf(ap[j]);
        }
        // ---------------- stage B tile (transposed into [n][k]) ----------------
        {
            int kr  = tid >> 3;      // 0..31
            int seg = tid & 7;       // 0..7 (8-col chunk)
            const float* bp = B + (size_t)(k0 + kr) * Nc + n0 + seg * 8;
            if (k0 + GTK < K)
                __builtin_prefetch((const void*)(bp + (size_t)GTK * Nc), 0, 0);
            #pragma unroll
            for (int j = 0; j < 8; ++j)
                Bs[(seg * 8 + j) * BSTR + kr] = f2bf(bp[j]);
        }
        __syncthreads();

        // ---------------- fragment loads (ds_load_b128 x8) + 4 WMMA ----------------
        FragU afrag[2], bfrag[2];
        #pragma unroll
        for (int i = 0; i < 2; ++i) {
            // ISA 16-bit A layout: lane row = lane&15, kb=(lane>>4)*8;
            // elems 0..7 -> K=kb..kb+7 ; elems 8..15 -> K=kb+16..kb+23
            int m  = wm * 32 + i * 16 + (lane & 15);
            int kb = (lane >> 4) * 8;
            afrag[i].u[0] = *(const uint4*)&As[m * ASTR + kb];
            afrag[i].u[1] = *(const uint4*)&As[m * ASTR + kb + 16];
        }
        #pragma unroll
        for (int j = 0; j < 2; ++j) {
            // B layout: lane col = lane&15, ks=(lane>>4)*8; elems 0..7 -> K=ks..ks+7 ; elems 8..15 -> K=ks+16..ks+23
            int n  = wn * 32 + j * 16 + (lane & 15);
            int ks = (lane >> 4) * 8;
            bfrag[j].u[0] = *(const uint4*)&Bs[n * BSTR + ks];
            bfrag[j].u[1] = *(const uint4*)&Bs[n * BSTR + ks + 16];
        }
        #pragma unroll
        for (int i = 0; i < 2; ++i)
            #pragma unroll
            for (int j = 0; j < 2; ++j)
                acc[i][j] = __builtin_amdgcn_wmma_f32_16x16x32_bf16(
                    false, afrag[i].v, false, bfrag[j].v,
                    (short)0, acc[i][j], false, false);
        __syncthreads();
    }

    // ---------------- epilogue ----------------
    // C/D layout: VGPR r -> row = r + (lane>=16 ? 8 : 0); col = lane&15
    #pragma unroll
    for (int i = 0; i < 2; ++i) {
        #pragma unroll
        for (int j = 0; j < 2; ++j) {
            int col  = n0 + wn * 32 + j * 16 + (lane & 15);
            int rowb = m0 + wm * 32 + i * 16 + ((lane >> 4) << 3);
            float bi = bias ? bias[col] : 0.0f;
            #pragma unroll
            for (int r = 0; r < 8; ++r) {
                int row = rowb + r;
                size_t idx = (size_t)row * Nc + col;
                float v = acc[i][j][r] + bi;
                v = act_apply(v, act);
                if (mulBuf) v *= mulBuf[idx];
                if (resBuf) v += resBuf[idx];
                if (accum)  Out[idx] += v;
                else        Out[idx]  = v;
            }
        }
    }
}

// ---------------------------------------------------------------------------
// Elementwise / row kernels
// ---------------------------------------------------------------------------
__global__ void fill_f32(float* __restrict__ p, int n, float v) {
    int i = blockIdx.x * 256 + threadIdx.x;
    if (i < n) p[i] = v;
}

// scatter-max: x_max[src[e], c] = max(h[dst[e], c]); values are relu'ed (>=0)
// so 0-init + integer atomicMax on IEEE bits == reference (isfinite->0 included)
__global__ void seg_max_kernel(const float* __restrict__ h,
                               const int* __restrict__ ei,
                               float* __restrict__ xmax) {
    int e = blockIdx.x;
    int c = threadIdx.x;
    int s = ei[e];
    int d = ei[EE + e];
    float val = h[(size_t)d * CC + c];
    atomicMax((int*)&xmax[(size_t)s * CC + c], __float_as_int(val));
}

__global__ void row_norm_kernel(const float* __restrict__ v, float* __restrict__ out) {
    int r = blockIdx.x, c = threadIdx.x;
    __shared__ float red[256];
    float x = v[(size_t)r * CC + c];
    red[c] = x * x; __syncthreads();
    for (int s = 128; s > 0; s >>= 1) { if (c < s) red[c] += red[c + s]; __syncthreads(); }
    float nrm = sqrtf(red[0]) + 1e-8f;
    out[(size_t)r * CC + c] = x / nrm;
}

__global__ void build_xcat_kernel(const float* __restrict__ x, const float* __restrict__ dv,
                                  const float* __restrict__ cv, const float* __restrict__ vn,
                                  float* __restrict__ xcat) {
    int r = blockIdx.x;
    for (int c = threadIdx.x; c < 4 * CC; c += 256) {
        const float* src; int cc;
        if      (c <     CC) { src = x;  cc = c;          }
        else if (c < 2 * CC) { src = dv; cc = c -     CC; }
        else if (c < 3 * CC) { src = cv; cc = c - 2 * CC; }
        else                 { src = vn; cc = c - 3 * CC; }
        xcat[(size_t)r * (4 * CC) + c] = src[(size_t)r * CC + cc];
    }
}

__device__ __forceinline__ float vcat_at(const float* v, const float* hod, const float* gx,
                                         int r, int c) {
    if (c <     CC) return v  [(size_t)r * CC + c];
    if (c < 2 * CC) return hod[(size_t)r * CC + c -     CC];
    return            gx [(size_t)r * CC + c - 2 * CC];
}

// vij = [v_cat, J(v_cat)] with v_cat=[v,hodge,g_x] (3C) and J = [-vc[384:], vc[:384]]
__global__ void build_vij_kernel(const float* __restrict__ v, const float* __restrict__ hod,
                                 const float* __restrict__ gx, float* __restrict__ vij) {
    int r = blockIdx.x;
    for (int c = threadIdx.x; c < 6 * CC; c += 256) {
        float val;
        if (c < 3 * CC) {
            val = vcat_at(v, hod, gx, r, c);
        } else {
            int jc = c - 3 * CC;                 // 0..767, half = 384
            if (jc < 384) val = -vcat_at(v, hod, gx, r, jc + 384);
            else          val =  vcat_at(v, hod, gx, r, jc - 384);
        }
        vij[(size_t)r * (6 * CC) + c] = val;
    }
}

// xs = layernorm( y + rowmax(y) ), y = glu/4
__global__ void ln_rowmax_kernel(const float* __restrict__ glu, const float* __restrict__ g,
                                 const float* __restrict__ b, float* __restrict__ out) {
    int r = blockIdx.x, c = threadIdx.x;
    __shared__ float red[256];
    float y = 0.25f * glu[(size_t)r * CC + c];
    red[c] = y; __syncthreads();
    for (int s = 128; s > 0; s >>= 1) { if (c < s) red[c] = fmaxf(red[c], red[c + s]); __syncthreads(); }
    float z = y + red[0];
    __syncthreads();
    red[c] = z; __syncthreads();
    for (int s = 128; s > 0; s >>= 1) { if (c < s) red[c] += red[c + s]; __syncthreads(); }
    float mean = red[0] * (1.0f / CC);
    __syncthreads();
    red[c] = z * z; __syncthreads();
    for (int s = 128; s > 0; s >>= 1) { if (c < s) red[c] += red[c + s]; __syncthreads(); }
    float var = red[0] * (1.0f / CC) - mean * mean;
    out[(size_t)r * CC + c] = (z - mean) * rsqrtf(var + 1e-5f) * g[c] + b[c];
}

// vs = y + rowmean(y), y = glu/4
__global__ void mean_add_kernel(const float* __restrict__ glu, float* __restrict__ out) {
    int r = blockIdx.x, c = threadIdx.x;
    __shared__ float red[256];
    float y = 0.25f * glu[(size_t)r * CC + c];
    red[c] = y; __syncthreads();
    for (int s = 128; s > 0; s >>= 1) { if (c < s) red[c] += red[c + s]; __syncthreads(); }
    float mean = red[0] * (1.0f / CC);
    out[(size_t)r * CC + c] = y + mean;
}

// ---------------------------------------------------------------------------
// Host orchestration
// ---------------------------------------------------------------------------
static inline void launch_gemm(hipStream_t s, const float* A, const float* B, float* O,
                               int M, int K, int Nc, int ta, const float* bias, int act,
                               const float* mul, const float* res, int accum) {
    dim3 grid(M / GTM, Nc / GTN), blk(256);
    gemm_wmma_bf16<<<grid, blk, 0, s>>>(A, B, O, M, K, Nc, ta, bias, act, mul, res, accum);
}

extern "C" void kernel_launch(void* const* d_in, const int* in_sizes, int n_in,
                              void* d_out, int out_size, void* d_ws, size_t ws_size,
                              hipStream_t stream) {
    (void)in_sizes; (void)n_in; (void)out_size; (void)ws_size;

    // ---- inputs (setup_inputs dict order, params flattened in insertion order) ----
    const float* x    = (const float*)d_in[0];
    const float* v    = (const float*)d_in[1];
    const float* grad = (const float*)d_in[2];
    const float* divm = (const float*)d_in[3];
    const int*   ei   = (const int*)  d_in[4];
    const float* wsm  = (const float*)d_in[5];
    const float* bsm  = (const float*)d_in[6];
    const float* ws   = (const float*)d_in[7];
    const float* bs   = (const float*)d_in[8];
    const float* wvm  = (const float*)d_in[9];
    const float* bvm  = (const float*)d_in[10];
    const float* ln_g = (const float*)d_in[11];
    const float* ln_b = (const float*)d_in[12];
    // as_: wq wk wv wo bq bk bv bo   (q/k are dead: softmax over seq-len 1 == 1)
    const float* as_wv = (const float*)d_in[15];
    const float* as_wo = (const float*)d_in[16];
    const float* as_bv = (const float*)d_in[19];
    const float* as_bo = (const float*)d_in[20];
    const float* av_wv = (const float*)d_in[23];
    const float* av_wo = (const float*)d_in[24];
    const float* av_bv = (const float*)d_in[27];
    const float* av_bo = (const float*)d_in[28];
    const float* gs_wval  = (const float*)d_in[29];  // [4,C,C]
    const float* gs_bval  = (const float*)d_in[30];  // [4,C]
    const float* gs_wgate = (const float*)d_in[31];
    const float* gs_bgate = (const float*)d_in[32];
    const float* gv_wval  = (const float*)d_in[33];
    const float* gv_bval  = (const float*)d_in[34];
    const float* gv_wgate = (const float*)d_in[35];
    const float* gv_bgate = (const float*)d_in[36];

    float* xs_out = (float*)d_out;        // [N,C]
    float* vs_out = xs_out + NC;          // [N,C]

    // ---- workspace layout (floats); total 22*NC*4 ≈ 88 MB ----
    float* W     = (float*)d_ws;
    float* h     = W;                     // [N,C]
    float* xmax  = W +  1 * (size_t)NC;
    float* divv  = W +  2 * (size_t)NC;
    float* curlv = W +  3 * (size_t)NC;
    float* vnorm = W +  4 * (size_t)NC;
    float* xcat  = W +  5 * (size_t)NC;   // [N,4C]
    float* t0    = W +  9 * (size_t)NC;
    float* t1    = W + 10 * (size_t)NC;
    float* t2    = W + 11 * (size_t)NC;
    float* gate  = W + 12 * (size_t)NC;
    float* glu   = W + 13 * (size_t)NC;
    float* hodge = W + 14 * (size_t)NC;
    float* gx    = W + 15 * (size_t)NC;
    float* vij   = W + 16 * (size_t)NC;   // [N,6C]

    const int actcodes[4] = {1, 2, 3, 4}; // relu, sigmoid, tanh, gelu

    // 1) h = relu(x @ wsm + bsm)
    launch_gemm(stream, x, wsm, h, NN, CC, CC, 0, bsm, 1, nullptr, nullptr, 0);
    // 2) x_max via scatter-max over edges (0-init; relu'ed values are >= 0)
    fill_f32<<<NC / 256, 256, 0, stream>>>(xmax, NC, 0.0f);
    seg_max_kernel<<<EE, CC, 0, stream>>>(h, ei, xmax);
    // 3) div_v = div @ v ; curl_v = div.T @ v
    launch_gemm(stream, divm, v, divv,  NN, NN, CC, 0, nullptr, 0, nullptr, nullptr, 0);
    launch_gemm(stream, divm, v, curlv, NN, NN, CC, 1, nullptr, 0, nullptr, nullptr, 0);
    // 4) v_norm
    row_norm_kernel<<<NN, CC, 0, stream>>>(v, vnorm);
    // 5) x_cat ; xs = x_max + relu(x_cat @ ws + bs)
    build_xcat_kernel<<<NN, 256, 0, stream>>>(x, divv, curlv, vnorm, xcat);
    launch_gemm(stream, xcat, ws, t0, NN, 4 * CC, CC, 0, bs, 1, nullptr, xmax, 0);
    // 6) attn as_ (degenerate): t2 = (t0 @ wv + bv) @ wo + bo
    launch_gemm(stream, t0, as_wv, t1, NN, CC, CC, 0, as_bv, 0, nullptr, nullptr, 0);
    launch_gemm(stream, t1, as_wo, t2, NN, CC, CC, 0, as_bo, 0, nullptr, nullptr, 0);
    // 7) GLU gs_: glu = sum_i act_i(t2@Wv_i+bv_i) * sigmoid(t2@Wg_i+bg_i)
    fill_f32<<<NC / 256, 256, 0, stream>>>(glu, NC, 0.0f);
    for (int i = 0; i < 4; ++i) {
        launch_gemm(stream, t2, gs_wgate + (size_t)i * CC * CC, gate,
                    NN, CC, CC, 0, gs_bgate + (size_t)i * CC, 2, nullptr, nullptr, 0);
        launch_gemm(stream, t2, gs_wval + (size_t)i * CC * CC, glu,
                    NN, CC, CC, 0, gs_bval + (size_t)i * CC, actcodes[i], gate, nullptr, 1);
    }
    // 8) xs = layernorm(glu/4 + rowmax(glu/4))  -> d_out (also feeds g_x GEMM)
    ln_rowmax_kernel<<<NN, CC, 0, stream>>>(glu, ln_g, ln_b, xs_out);
    // 9) hodge = grad @ div_v ; g_x = grad @ xs
    launch_gemm(stream, grad, divv,   hodge, NN, NN, CC, 0, nullptr, 0, nullptr, nullptr, 0);
    launch_gemm(stream, grad, xs_out, gx,    NN, NN, CC, 0, nullptr, 0, nullptr, nullptr, 0);
    // 10) vij = [v_cat, J(v_cat)] ; vs = relu(vij @ wvm + bvm)
    build_vij_kernel<<<NN, 256, 0, stream>>>(v, hodge, gx, vij);
    launch_gemm(stream, vij, wvm, t0, NN, 6 * CC, CC, 0, bvm, 1, nullptr, nullptr, 0);
    // 11) attn av_ (degenerate)
    launch_gemm(stream, t0, av_wv, t1, NN, CC, CC, 0, av_bv, 0, nullptr, nullptr, 0);
    launch_gemm(stream, t1, av_wo, t2, NN, CC, CC, 0, av_bo, 0, nullptr, nullptr, 0);
    // 12) GLU gv_
    fill_f32<<<NC / 256, 256, 0, stream>>>(glu, NC, 0.0f);
    for (int i = 0; i < 4; ++i) {
        launch_gemm(stream, t2, gv_wgate + (size_t)i * CC * CC, gate,
                    NN, CC, CC, 0, gv_bgate + (size_t)i * CC, 2, nullptr, nullptr, 0);
        launch_gemm(stream, t2, gv_wval + (size_t)i * CC * CC, glu,
                    NN, CC, CC, 0, gv_bval + (size_t)i * CC, actcodes[i], gate, nullptr, 1);
    }
    // 13) vs = glu/4 + rowmean(glu/4)
    mean_add_kernel<<<NN, CC, 0, stream>>>(glu, vs_out);
}
